// W_Conv_12970801234201
// MI455X (gfx1250) — matrix operand order repaired and checked by reading, not compiled, not run on hardware
//
#include <hip/hip_runtime.h>

// SCNN bidirectional width recurrence for MI455X (gfx1250, wave32, WMMA).
// One persistent workgroup per batch; recurrent column state lives in LDS
// (double-buffered bf16); per-step GEMM (M=128, N=72->80, K=1152) done with
// v_wmma_f32_16x16x32_bf16. Weights pre-converted to bf16 in d_ws.

#define Bn 16
#define Cn 128
#define Hn 72
#define Wn 200
#define Kn 9
#define KKn (Cn * Kn)      // 1152 = im2col K
#define KSTEPS (KKn / 32)  // 36 WMMA k-steps
#define NTILE 5            // 80 = 5 * 16 h-columns (72 valid)
#define HP 96              // padded h rows in LDS (valid data at 4..75, reads reach 87)
#define ROWP 136           // padded row stride in halves (272B, bank-friendly)

typedef __attribute__((ext_vector_type(16))) __bf16       v16bf;
typedef __attribute__((ext_vector_type(8)))  float        v8f;
typedef __attribute__((ext_vector_type(4)))  unsigned int v4u;

union Frag {
    v4u   u4[2];
    v16bf bf;
};

__device__ __forceinline__ unsigned short f2bf(float f) {
    unsigned int u = __float_as_uint(f);
    u += 0x7FFFu + ((u >> 16) & 1u);   // round-to-nearest-even
    return (unsigned short)(u >> 16);
}

// One recurrence step: newcol = src_col(i) + relu(W * xprev + bias),
// write fp32 to dst, bf16 state into xnew. Single barrier (double-buffered LDS).
__device__ __forceinline__ void scnn_step(
    int b, int i, int ipf,
    const float* __restrict__ src, float* __restrict__ dst,
    const unsigned short* __restrict__ wrow,   // this lane's A row (co = 16*mi + lo)
    const float* bias,                         // 8 per-lane bias regs
    const unsigned short* __restrict__ xprev,  // LDS: previous column (bf16)
    unsigned short* __restrict__ xnew,         // LDS: next column buffer
    int mi, int lo, int lhi)
{
    v8f acc[NTILE] = {};

    for (int ks = 0; ks < KSTEPS; ++ks) {
        const int kk = ks * 32;
        const int kr = kk >> 7;        // conv tap (0..8): constant h shift
        const int cb = kk & (Cn - 1);  // ci base within tap

        // A fragment (16x32 bf16): lane holds K = {lhi*8..+7, 16+lhi*8..+7}
        Frag fa;
        const v4u* pa = (const v4u*)(wrow + kk + lhi * 8);
        fa.u4[0] = pa[0];
        fa.u4[1] = pa[2];

#pragma unroll
        for (int n = 0; n < NTILE; ++n) {
            // B fragment (32x16 bf16): lane (n-col = lo) holds K = {lhi*8..+7, 16+lhi*8..+7},
            // K maps to ci = cb + K; h index shifted by tap kr.
            Frag fb;
            const v4u* pb =
                (const v4u*)(xprev + (n * 16 + lo + kr) * ROWP + cb + lhi * 8);
            fb.u4[0] = pb[0];
            fb.u4[1] = pb[2];
            acc[n] = __builtin_amdgcn_wmma_f32_16x16x32_bf16(
                false, fa.bf, false, fb.bf, (short)0, acc[n], false, false);
        }
    }

    // Epilogue: D layout -> co = 16*mi + 8*lhi + v, h = 16*n + lo
#pragma unroll
    for (int n = 0; n < NTILE; ++n) {
        const int h = n * 16 + lo;
        if (h < Hn) {
#pragma unroll
            for (int v = 0; v < 8; ++v) {
                const int co = mi * 16 + lhi * 8 + v;
                const int gi = ((b * Cn + co) * Hn + h) * Wn + i;
                float y = acc[n][v] + bias[v];
                y = fmaxf(y, 0.0f);
                const float r = src[gi] + y;
                dst[gi] = r;
                xnew[(h + 4) * ROWP + co] = f2bf(r);
                __builtin_prefetch(&src[gi - i + ipf], 0, 3);
            }
        }
    }
    __syncthreads();
}

__global__ void __launch_bounds__(256, 1)
scnn_kernel(const float* __restrict__ xin,
            const unsigned short* __restrict__ wR,
            const unsigned short* __restrict__ wL,
            const float* __restrict__ bR,
            const float* __restrict__ bL,
            float* __restrict__ out)
{
    __shared__ unsigned short xlds[2 * HP * ROWP];  // 2 x 25.5KB bf16 column buffers

    const int b    = blockIdx.x;
    const int tid  = threadIdx.x;
    const int mi   = tid >> 5;        // wave id = M tile
    const int lane = tid & 31;
    const int lo   = lane & 15;
    const int lhi  = lane >> 4;

    // Zero both LDS buffers (establishes conv zero-padding halos).
    for (int t = tid; t < 2 * HP * ROWP; t += 256) xlds[t] = 0;

    // Load column 0 into buffer 0; column 0 passes through unchanged.
    for (int t = tid; t < Cn * Hn; t += 256) {
        const int ci = t / Hn, h = t - ci * Hn;
        const int gi = ((b * Cn + ci) * Hn + h) * Wn;  // i = 0
        const float v = xin[gi];
        out[gi] = v;
        xlds[(h + 4) * ROWP + ci] = f2bf(v);
    }

    float biasR[8], biasL[8];
#pragma unroll
    for (int v = 0; v < 8; ++v) {
        const int co = mi * 16 + lhi * 8 + v;
        biasR[v] = bR[co];
        biasL[v] = bL[co];
    }
    __syncthreads();

    const unsigned short* wrowR = wR + (mi * 16 + lo) * KKn;
    const unsigned short* wrowL = wL + (mi * 16 + lo) * KKn;

    int p = 0;
    // Rightward pass: i = 1 .. 199
    for (int i = 1; i < Wn; ++i) {
        scnn_step(b, i, (i + 1 < Wn) ? i + 1 : i, xin, out, wrowR, biasR,
                  xlds + p * HP * ROWP, xlds + (p ^ 1) * HP * ROWP, mi, lo, lhi);
        p ^= 1;
    }
    // Leftward pass: i = 198 .. 1, reads cols_r back from `out`, updates in place.
    for (int i = Wn - 2; i >= 1; --i) {
        scnn_step(b, i, (i - 1 >= 1) ? i - 1 : i, out, out, wrowL, biasL,
                  xlds + p * HP * ROWP, xlds + (p ^ 1) * HP * ROWP, mi, lo, lhi);
        p ^= 1;
    }
}

// Convert both weight tensors [C,C,K] f32 -> bf16 with layout [co][kr*128+ci]
// so each 32-wide WMMA k-step stays inside a single conv tap kr.
__global__ void cvt_w_kernel(const float* __restrict__ wr,
                             const float* __restrict__ wl,
                             unsigned short* __restrict__ dst)
{
    const int NW  = Cn * Cn * Kn;  // 147456
    const int idx = blockIdx.x * 256 + threadIdx.x;
    if (idx >= 2 * NW) return;
    const float* srcp = (idx < NW) ? wr : wl;
    const int j  = (idx < NW) ? idx : idx - NW;
    const int co = j / KKn;
    const int kk = j - co * KKn;
    const int kr = kk >> 7;
    const int ci = kk & (Cn - 1);
    dst[idx] = f2bf(srcp[(co * Cn + ci) * Kn + kr]);
}

extern "C" void kernel_launch(void* const* d_in, const int* in_sizes, int n_in,
                              void* d_out, int out_size, void* d_ws, size_t ws_size,
                              hipStream_t stream) {
    const float* x  = (const float*)d_in[0];
    const float* wr = (const float*)d_in[1];
    const float* br = (const float*)d_in[2];
    const float* wl = (const float*)d_in[3];
    const float* bl = (const float*)d_in[4];
    float* out = (float*)d_out;

    unsigned short* wbf = (unsigned short*)d_ws;  // needs 2*147456*2 = 576 KB scratch
    const int NW = Cn * Cn * Kn;

    cvt_w_kernel<<<(2 * NW + 255) / 256, 256, 0, stream>>>(wr, wl, wbf);
    scnn_kernel<<<Bn, 256, 0, stream>>>(x, wbf, wbf + NW, br, bl, out);
}
